// ODEGRU_89953795047636
// MI455X (gfx1250) — matrix operand order repaired and checked by reading, not compiled, not run on hardware
//
#include <hip/hip_runtime.h>

typedef _Float16 f16;
typedef __attribute__((ext_vector_type(16))) _Float16 v16h;
typedef __attribute__((ext_vector_type(8)))  _Float16 v8h;
typedef __attribute__((ext_vector_type(8)))  float    v8f;
typedef __attribute__((ext_vector_type(4)))  float    v4f;

#define BB 128
#define LL 2048
#define II 64
#define DD 64

static __device__ __forceinline__ v8f wmma16(v16h a, v16h b, v8f c) {
  return __builtin_amdgcn_wmma_f32_16x16x32_f16(false, a, false, b, (short)0, c,
                                                false, false);
}

static __device__ __forceinline__ v8f splat8(float v) {
  v8f r;
#pragma unroll
  for (int i = 0; i < 8; ++i) r[i] = v;
  return r;
}

// Workgroup barrier that only fences LDS (dscnt), not global loads/stores.
// __syncthreads() would add s_wait_storecnt/loadcnt 0, putting the fire-and-
// forget global 'out' stores and the prefetched x loads on the critical path.
static __device__ __forceinline__ void wg_barrier_lds() {
  asm volatile("s_wait_dscnt 0x0\n\t"
               "s_barrier_signal -1\n\t"
               "s_barrier_wait -1" ::: "memory");
}

// ---- branchless native transcendentals ----
static __device__ __forceinline__ float fast_exp(float x) {
  return __builtin_amdgcn_exp2f(x * 1.44269504088896340736f);  // v_exp_f32
}
static __device__ __forceinline__ float fast_rcp(float x) {
  return __builtin_amdgcn_rcpf(x);                              // v_rcp_f32
}
static __device__ __forceinline__ float fast_sigmoid(float x) {
  return fast_rcp(1.0f + fast_exp(-x));
}
static __device__ __forceinline__ float fast_tanh(float x) {
#if __has_builtin(__builtin_amdgcn_tanhf)
  return __builtin_amdgcn_tanhf(x);                             // v_tanh_f32
#else
  return 1.0f - 2.0f * fast_rcp(1.0f + fast_exp(2.0f * x));
#endif
}

// Convert 16 raw floats (A-fragment pattern) to f16 with scale.
static __device__ __forceinline__ v16h cvt_afrag(v4f q0, v4f q1, v4f q2, v4f q3,
                                                 float s) {
  v16h a;
#pragma unroll
  for (int i = 0; i < 4; ++i) {
    a[i]      = (f16)(q0[i] * s);
    a[i + 4]  = (f16)(q1[i] * s);
    a[i + 8]  = (f16)(q2[i] * s);
    a[i + 12] = (f16)(q3[i] * s);
  }
  return a;
}

static __device__ __forceinline__ v16h load_rowpair_f16(const float* __restrict__ p,
                                                        float s) {
  return cvt_afrag(*(const v4f*)(p + 0), *(const v4f*)(p + 4),
                   *(const v4f*)(p + 16), *(const v4f*)(p + 20), s);
}

// B-fragment: floats [0,8) and [16,24) of one weight row -> f16
static __device__ __forceinline__ v16h load_bfrag(const float* __restrict__ p) {
  v4f q0 = *(const v4f*)(p + 0);
  v4f q1 = *(const v4f*)(p + 4);
  v4f q2 = *(const v4f*)(p + 16);
  v4f q3 = *(const v4f*)(p + 20);
  v16h b;
#pragma unroll
  for (int i = 0; i < 4; ++i) {
    b[i]      = (f16)q0[i];
    b[i + 4]  = (f16)q1[i];
    b[i + 8]  = (f16)q2[i];
    b[i + 12] = (f16)q3[i];
  }
  return b;
}

// A-fragment from a 16x64 f16 matrix staged in LDS (two ds_load_b128 per lane)
static __device__ __forceinline__ v16h lds_afrag(const f16* base, int m, int g,
                                                 int c) {
  const f16* p = base + m * 64 + c * 32 + g * 8;
  v8h lo = *(const v8h*)(p);
  v8h hi = *(const v8h*)(p + 16);
  return __builtin_shufflevector(lo, hi, 0, 1, 2, 3, 4, 5, 6, 7, 8, 9, 10, 11,
                                 12, 13, 14, 15);
}

__global__ __launch_bounds__(128)
void odegru_scan(const float* __restrict__ x,    // (B,L,I)
                 const float* __restrict__ td,   // (B,L)
                 const int*   __restrict__ seq_lens,
                 const float* __restrict__ h0,   // (D)
                 const float* __restrict__ w_ih, // (3D,I)
                 const float* __restrict__ w_hh, // (3D,D)
                 const float* __restrict__ b_ih,
                 const float* __restrict__ b_hh,
                 const float* __restrict__ dw1,  // (D,D)
                 const float* __restrict__ db1,
                 const float* __restrict__ dw2,  // (D,D)
                 const float* __restrict__ db2,
                 float* __restrict__ out)        // (B,L,D)
{
  __shared__ __align__(16) f16 s0[16 * 64];  // tanh(h@dw1^T) staging
  __shared__ __align__(16) f16 s1[16 * 64];  // h_ode staging
  __shared__ __align__(16) f16 s2[16 * 64];  // h_new staging

  const int tid  = threadIdx.x;
  const int wv   = tid >> 5;         // wave 0..3 -> owns 16-col slice
  const int lane = tid & 31;
  const int g    = lane >> 4;        // half-wave group
  const int ln   = lane & 15;
  const int bm   = blockIdx.x * 16;  // batch-row base of this block
  const int col  = wv * 16 + ln;     // column within D (0..63)

  // ---- resident weight B-fragments (w_ih/w_hh: 3 gate tiles x 2 K-chunks) ----
  v16h wihf[3][2], whhf[3][2], dw1f[2], dw2f[2];
#pragma unroll
  for (int gi = 0; gi < 3; ++gi) {
#pragma unroll
    for (int c = 0; c < 2; ++c) {
      wihf[gi][c] = load_bfrag(w_ih + (gi * 64 + col) * II + c * 32 + g * 8);
      whhf[gi][c] = load_bfrag(w_hh + (gi * 64 + col) * DD + c * 32 + g * 8);
    }
  }
#pragma unroll
  for (int c = 0; c < 2; ++c) {
    dw1f[c] = load_bfrag(dw1 + col * DD + c * 32 + g * 8);
    dw2f[c] = load_bfrag(dw2 + col * DD + c * 32 + g * 8);
  }

  // ---- biases (r/z may pre-combine; n-gate must stay split) ----
  const float bias_r  = b_ih[col]       + b_hh[col];
  const float bias_z  = b_ih[64 + col]  + b_hh[64 + col];
  const float bias_nx = b_ih[128 + col];
  const float bias_nh = b_hh[128 + col];
  const float bias_d1 = db1[col];
  const float bias_d2 = db2[col];

  // ---- sequence lengths ----
  const int sl_a = seq_lens[bm + ln];  // row mask for A-fragments (row = ln)
  int sl_c[8];
#pragma unroll
  for (int r = 0; r < 8; ++r) sl_c[r] = seq_lens[bm + g * 8 + r];  // C-layout rows

  // ---- h init: broadcast h0 to all rows ----
  v16h hA[2];
  hA[0] = load_rowpair_f16(h0 + 0 * 32 + g * 8, 1.0f);
  hA[1] = load_rowpair_f16(h0 + 1 * 32 + g * 8, 1.0f);
  float h_ct[8];
  const float h0c = h0[col];
#pragma unroll
  for (int r = 0; r < 8; ++r) h_ct[r] = h0c;

  const float* xp_lane = x + (size_t)(bm + ln) * LL * II + g * 8;
  const float* td_lane = td + (size_t)(bm + g * 8) * LL;  // + r*LL + t

  for (int t = 0; t < LL; ++t) {
    // ======== issue-early loads (consumed late; latency hidden by MLP) ========
    const float* xp = xp_lane + (size_t)t * II;
    v4f xq0 = *(const v4f*)(xp + 0);
    v4f xq1 = *(const v4f*)(xp + 4);
    v4f xq2 = *(const v4f*)(xp + 16);
    v4f xq3 = *(const v4f*)(xp + 20);
    v4f xq4 = *(const v4f*)(xp + 32);
    v4f xq5 = *(const v4f*)(xp + 36);
    v4f xq6 = *(const v4f*)(xp + 48);
    v4f xq7 = *(const v4f*)(xp + 52);
    float dtv[8];
#pragma unroll
    for (int r = 0; r < 8; ++r) dtv[r] = td_lane[(size_t)r * LL + t];

    // ======== diff layer 1: tanh(h @ dw1^T + db1) ========
    v8f c1 = splat8(bias_d1);
    c1 = wmma16(hA[0], dw1f[0], c1);
    c1 = wmma16(hA[1], dw1f[1], c1);
#pragma unroll
    for (int r = 0; r < 8; ++r)
      s0[(g * 8 + r) * 64 + col] = (f16)fast_tanh(c1[r]);
    wg_barrier_lds();
    v16h tA0 = lds_afrag(s0, ln, g, 0);
    v16h tA1 = lds_afrag(s0, ln, g, 1);

    // ======== diff layer 2 + Euler: h_ode = h + (t1 @ dw2^T + db2) * dt ========
    v8f c2 = splat8(bias_d2);
    c2 = wmma16(tA0, dw2f[0], c2);
    c2 = wmma16(tA1, dw2f[1], c2);
    float hode[8];
#pragma unroll
    for (int r = 0; r < 8; ++r) {
      const float dtm = (t < sl_c[r]) ? dtv[r] : 0.0f;
      hode[r] = h_ct[r] + c2[r] * dtm;
      s1[(g * 8 + r) * 64 + col] = (f16)hode[r];
    }
    // convert x while the s1 ds_stores drain (x loads have had the whole
    // first half of the step to complete)
    const float xm = (t < sl_a) ? 1.0f : 0.0f;
    v16h xA0 = cvt_afrag(xq0, xq1, xq2, xq3, xm);
    v16h xA1 = cvt_afrag(xq4, xq5, xq6, xq7, xm);
    wg_barrier_lds();
    v16h oA0 = lds_afrag(s1, ln, g, 0);
    v16h oA1 = lds_afrag(s1, ln, g, 1);

    // ======== gates: r/z fuse gx+gh in one chain; n-gate stays split ========
    v8f gr = splat8(bias_r);
    gr = wmma16(xA0, wihf[0][0], gr); gr = wmma16(xA1, wihf[0][1], gr);
    gr = wmma16(oA0, whhf[0][0], gr); gr = wmma16(oA1, whhf[0][1], gr);
    v8f gz = splat8(bias_z);
    gz = wmma16(xA0, wihf[1][0], gz); gz = wmma16(xA1, wihf[1][1], gz);
    gz = wmma16(oA0, whhf[1][0], gz); gz = wmma16(oA1, whhf[1][1], gz);
    v8f gxn = splat8(bias_nx);
    gxn = wmma16(xA0, wihf[2][0], gxn); gxn = wmma16(xA1, wihf[2][1], gxn);
    v8f gn = splat8(bias_nh);
    gn = wmma16(oA0, whhf[2][0], gn); gn = wmma16(oA1, whhf[2][1], gn);

    // ======== blend + store (no conditionals in the hot loop) ========
#pragma unroll
    for (int r = 0; r < 8; ++r) {
      const int row = g * 8 + r;
      const float rr = fast_sigmoid(gr[r]);
      const float zz = fast_sigmoid(gz[r]);
      const float nn = fast_tanh(gxn[r] + rr * gn[r]);
      const float hn = (1.0f - zz) * nn + zz * hode[r];
      h_ct[r] = hn;
      out[((size_t)(bm + row) * LL + t) * DD + col] = hn;
      s2[row * 64 + col] = (f16)hn;
    }
    wg_barrier_lds();
    hA[0] = lds_afrag(s2, ln, g, 0);
    hA[1] = lds_afrag(s2, ln, g, 1);
  }
}

// Epilogue: final[b] = outputs[b, seq_len[b]-1, :]
__global__ __launch_bounds__(64)
void gather_final(const float* __restrict__ out, const int* __restrict__ seq_lens,
                  float* __restrict__ fin) {
  const int b = blockIdx.x;
  const int d = threadIdx.x;
  const int t = seq_lens[b] - 1;
  fin[(size_t)b * DD + d] = out[((size_t)b * LL + t) * DD + d];
}

extern "C" void kernel_launch(void* const* d_in, const int* in_sizes, int n_in,
                              void* d_out, int out_size, void* d_ws,
                              size_t ws_size, hipStream_t stream) {
  (void)in_sizes; (void)n_in; (void)out_size; (void)d_ws; (void)ws_size;
  const float* x    = (const float*)d_in[0];
  const float* td   = (const float*)d_in[1];
  const int*   sl   = (const int*)d_in[2];
  const float* h0   = (const float*)d_in[3];
  const float* wih  = (const float*)d_in[4];
  const float* whh  = (const float*)d_in[5];
  const float* bih  = (const float*)d_in[6];
  const float* bhh  = (const float*)d_in[7];
  const float* dw1p = (const float*)d_in[8];
  const float* db1p = (const float*)d_in[9];
  const float* dw2p = (const float*)d_in[10];
  const float* db2p = (const float*)d_in[11];

  float* out = (float*)d_out;                       // (B,L,D)
  float* fin = out + (size_t)BB * LL * DD;          // (1,B,D)

  odegru_scan<<<BB / 16, 128, 0, stream>>>(x, td, sl, h0, wih, whh, bih, bhh,
                                           dw1p, db1p, dw2p, db2p, out);
  gather_final<<<BB, DD, 0, stream>>>(out, sl, fin);
}
